// ScaledDotProductAttention_12137577578859
// MI455X (gfx1250) — matrix operand order repaired and checked by reading, not compiled, not run on hardware
//
#include <hip/hip_runtime.h>

typedef __bf16 bfx16 __attribute__((ext_vector_type(16)));
typedef __bf16 bfx8  __attribute__((ext_vector_type(8)));
typedef float  fx8   __attribute__((ext_vector_type(8)));

#define ATT_LOG2E 1.4426950408889634f

constexpr int AB = 2, AH = 16, AS = 2048, AD = 64;

// ---- prep: qb = q * (1/8) * log2(e)  (scores come out in exp2 domain) ------
__global__ void attn_cvt_qk_12137(const float* __restrict__ q,
                                  const float* __restrict__ k,
                                  __bf16* __restrict__ qb,
                                  __bf16* __restrict__ kb, int n) {
  int i = blockIdx.x * blockDim.x + threadIdx.x;
  if (i < n) {
    qb[i] = (__bf16)(q[i] * (0.125f * ATT_LOG2E));
    kb[i] = (__bf16)k[i];
  }
}

// ---- prep: v[bh][s][d] -> vt[bh][d][s] bf16 --------------------------------
__global__ void attn_cvt_v_12137(const float* __restrict__ v,
                                 __bf16* __restrict__ vt) {
  int s  = blockIdx.x * blockDim.x + threadIdx.x;
  int d  = blockIdx.y;
  int bh = blockIdx.z;
  vt[((size_t)bh * AD + d) * AS + s] = (__bf16)v[((size_t)bh * AS + s) * AD + d];
}

// ---- main: one wave per 16-query tile --------------------------------------
__global__ __launch_bounds__(32) void attn_main_12137(
    const float* __restrict__ newmask, const unsigned char* __restrict__ mask,
    const __bf16* __restrict__ qb, const __bf16* __restrict__ kb,
    const __bf16* __restrict__ vt, float* __restrict__ out,
    float* __restrict__ attn) {
  __shared__ __bf16 ptile[16 * 32];  // P chunk (16 queries x 32 keys), bf16

  const int lane = threadIdx.x;
  const int lo = lane & 15;
  const int hi = lane >> 4;
  const int bh = blockIdx.y;
  const int b  = bh >> 4;  // H == 16
  const int q0 = blockIdx.x * 16;

  // Q A-fragments: lane row = lo; lanes 0-15 K=0..15, lanes 16-31
  // K=16..31 (kbase = hi*16); two fragments cover D=64.
  const __bf16* qrow = qb + ((size_t)bh * AS + (q0 + lo)) * AD;
  bfx16 aq[2];
#pragma unroll
  for (int f = 0; f < 2; ++f) {
    int db = f * 32 + hi * 16;
    *(bfx8*)&aq[f]       = *(const bfx8*)(qrow + db);
    *((bfx8*)&aq[f] + 1) = *(const bfx8*)(qrow + db + 8);
  }

  const unsigned char* mbase =
      mask + ((size_t)b * AS + (q0 + hi * 8)) * AS + lo;

  // raw score tile (exp2 domain) for k-tile kt; accum: m = v+8*hi, n = lo
  auto compute_scores = [&](int kt, fx8& c) {
    const __bf16* krow = kb + ((size_t)bh * AS + (kt * 16 + lo)) * AD;
    bfx16 bk0, bk1;
    *(bfx8*)&bk0       = *(const bfx8*)(krow + hi * 16);
    *((bfx8*)&bk0 + 1) = *(const bfx8*)(krow + hi * 16 + 8);
    *(bfx8*)&bk1       = *(const bfx8*)(krow + 32 + hi * 16);
    *((bfx8*)&bk1 + 1) = *(const bfx8*)(krow + 32 + hi * 16 + 8);
    c = (fx8){0.f, 0.f, 0.f, 0.f, 0.f, 0.f, 0.f, 0.f};
    c = __builtin_amdgcn_wmma_f32_16x16x32_bf16(false, aq[0], false, bk0,
                                                (short)0, c, false, false);
    c = __builtin_amdgcn_wmma_f32_16x16x32_bf16(false, aq[1], false, bk1,
                                                (short)0, c, false, false);
  };

  // ---- pass 1: row sums of exp2(score); masked elements contribute 0 ----
  float sm[8];
#pragma unroll
  for (int v = 0; v < 8; ++v) sm[v] = 0.f;

  for (int kt = 0; kt < AS / 16; ++kt) {
    fx8 c;
    compute_scores(kt, c);
    const unsigned char* mrow = mbase + (size_t)kt * 16;
#pragma unroll
    for (int v = 0; v < 8; ++v) {
      float e = __builtin_amdgcn_exp2f(c[v]);
      sm[v] += mrow[(size_t)v * AS] ? 0.f : e;
    }
  }
#pragma unroll
  for (int v = 0; v < 8; ++v) {
#pragma unroll
    for (int off = 1; off < 16; off <<= 1)
      sm[v] += __shfl_xor(sm[v], off, 32);
    sm[v] = 1.0f / sm[v];  // keep inverse of row sum
  }

  // ---- pass 2: recompute, emit attn = e*new_mask*inv, accumulate P@V ----
  fx8 o[4];
#pragma unroll
  for (int dt = 0; dt < 4; ++dt)
    o[dt] = (fx8){0.f, 0.f, 0.f, 0.f, 0.f, 0.f, 0.f, 0.f};

  const float* nmbase = newmask + ((size_t)b * AS + q0) * AS;
  float* abase        = attn + ((size_t)bh * AS + q0) * AS;
  const __bf16* vbase = vt + (size_t)bh * AD * AS;  // [d][s]

  for (int j = 0; j < AS / 32; ++j) {
#pragma unroll
    for (int t2 = 0; t2 < 2; ++t2) {
      int kt = 2 * j + t2;
      fx8 c;
      compute_scores(kt, c);
      const unsigned char* mrow = mbase + (size_t)kt * 16;
#pragma unroll
      for (int v = 0; v < 8; ++v) {
        int m = v + hi * 8;
        size_t idx = (size_t)m * AS + (size_t)kt * 16 + lo;
        float e = __builtin_amdgcn_exp2f(c[v]);
        // z = unnormalized attn incl. post-softmax mask-mul (0 if masked)
        float z = mrow[(size_t)v * AS]
                      ? 0.f
                      : e * __builtin_nontemporal_load(nmbase + idx);
        __builtin_nontemporal_store(z * sm[v], abase + idx);  // streaming 512MB
        ptile[m * 32 + t2 * 16 + lo] = (__bf16)z;  // transpose via LDS
      }
    }
    // P A-fragment (16 queries x 32 keys) from LDS in A layout
    bfx16 pa;
    {
      const __bf16* prow = &ptile[lo * 32 + hi * 16];
      *(bfx8*)&pa       = *(const bfx8*)(prow);
      *((bfx8*)&pa + 1) = *(const bfx8*)(prow + 8);
    }
    // V B-fragments: lane column = output d (dt*16+lo), 16 contiguous keys
    // per half-wave from transposed vt.
    int key0 = j * 32 + hi * 16;
#pragma unroll
    for (int dt = 0; dt < 4; ++dt) {
      const __bf16* vrow = vbase + (size_t)(dt * 16 + lo) * AS + key0;
      bfx16 bv;
      *(bfx8*)&bv       = *(const bfx8*)(vrow);
      *((bfx8*)&bv + 1) = *(const bfx8*)(vrow + 8);
      o[dt] = __builtin_amdgcn_wmma_f32_16x16x32_bf16(false, pa, false, bv,
                                                      (short)0, o[dt], false,
                                                      false);
    }
  }

  // ---- write output, applying deferred 1/rowsum ----
  float* orow = out + ((size_t)bh * AS + q0) * AD;
#pragma unroll
  for (int dt = 0; dt < 4; ++dt)
#pragma unroll
    for (int v = 0; v < 8; ++v)
      orow[(size_t)(v + hi * 8) * AD + dt * 16 + lo] = o[dt][v] * sm[v];
}

// ---------------------------------------------------------------------------
extern "C" void kernel_launch(void* const* d_in, const int* in_sizes, int n_in,
                              void* d_out, int out_size, void* d_ws,
                              size_t ws_size, hipStream_t stream) {
  (void)in_sizes; (void)n_in; (void)out_size; (void)ws_size;
  const float* q  = (const float*)d_in[0];
  const float* k  = (const float*)d_in[1];
  const float* v  = (const float*)d_in[2];
  const float* nm = (const float*)d_in[3];
  const unsigned char* mask = (const unsigned char*)d_in[4];  // numpy bool

  const size_t nqk = (size_t)AB * AH * AS * AD;  // 4M elements
  __bf16* qb = (__bf16*)d_ws;
  __bf16* kb = qb + nqk;
  __bf16* vt = kb + nqk;

  float* out  = (float*)d_out;
  float* attn = out + nqk;  // outputs concatenated: output then attn

  attn_cvt_qk_12137<<<(int)((nqk + 255) / 256), 256, 0, stream>>>(q, k, qb, kb,
                                                                  (int)nqk);
  attn_cvt_v_12137<<<dim3(AS / 256, AD, AB * AH), 256, 0, stream>>>(v, vt);
  attn_main_12137<<<dim3(AS / 16, AB * AH), 32, 0, stream>>>(nm, mask, qb, kb,
                                                             vt, out, attn);
}
